// TaskAlignedAssignerMutual_26542897889709
// MI455X (gfx1250) — matrix operand order repaired and checked by reading, not compiled, not run on hardware
//
#include <hip/hip_runtime.h>
#include <hip/hip_bf16.h>
#include <stdint.h>

// ---------------------------------------------------------------------------
// Problem constants (from reference)
// ---------------------------------------------------------------------------
#define BSZ   8
#define NA    21504
#define NG    48
#define NC    80
#define NLUT  401
#define TOPKK 13
#define EPSF  1e-9f

// Output layout (floats, concatenated in reference return order)
static constexpr size_t O_LBL    = 0;                          // (B,A)
static constexpr size_t O_BOX    = O_LBL + (size_t)BSZ * NA;   // (B,A,4)
static constexpr size_t O_ANG    = O_BOX + (size_t)BSZ * NA * 4;
static constexpr size_t O_SCORES = O_ANG + (size_t)BSZ * NA;   // (B,A,80)
static constexpr size_t O_FG     = O_SCORES + (size_t)BSZ * NA * NC;
static constexpr size_t O_NT     = O_FG + (size_t)BSZ * NA;    // (B,G,A)
static constexpr size_t O_NS     = O_NT + (size_t)BSZ * NG * NA;
static constexpr size_t O_MA     = O_NS + (size_t)BSZ * NG * NA;

// Workspace layout (floats)
static constexpr size_t W_MET = 0;                                  // align_t * mask_in (topk metric)
static constexpr size_t W_AS  = W_MET + (size_t)BSZ * NG * NA;      // align_s
static constexpr size_t W_AT  = W_AS  + (size_t)BSZ * NG * NA;      // align_t
static constexpr size_t W_OS  = W_AT  + (size_t)BSZ * NG * NA;      // overlaps_s
static constexpr size_t W_OT  = W_OS  + (size_t)BSZ * NG * NA;      // overlaps_t
static constexpr size_t W_MIN = W_OT  + (size_t)BSZ * NG * NA;      // mask_in_gts
static constexpr size_t W_MP  = W_MIN + (size_t)BSZ * NG * NA;      // mask_pos (final)
static constexpr size_t W_THR = W_MP  + (size_t)BSZ * NG * NA;      // (B,G) topk thresholds
static constexpr size_t W_RS  = W_THR + (size_t)BSZ * NG;           // (B,G,6) row maxima

// ---------------------------------------------------------------------------
// CDNA5 helpers
// ---------------------------------------------------------------------------
typedef __attribute__((ext_vector_type(16))) _Float16 v16h;
typedef __attribute__((ext_vector_type(8)))  float    v8f;
typedef __attribute__((ext_vector_type(4)))  unsigned v4u;
typedef __attribute__((ext_vector_type(8)))  unsigned v8u;

__device__ __forceinline__ unsigned lds_offset(const void* p) {
  // Generic (flat) pointers to LDS carry the LDS byte offset in bits [31:0].
  return (unsigned)(uintptr_t)p;
}

// 1-D Tensor Data Mover load: N contiguous 4-byte elements, global -> LDS.
// Per-wave instruction (scalar operands); tracked with TENSORcnt.
__device__ __forceinline__ void tdm_load_1d(const void* gaddr, unsigned lds, unsigned n) {
  unsigned long long ga = (unsigned long long)(uintptr_t)gaddr;
  // D# group 0: [1:0]=count=1, [63:32]=lds_addr, [120:64]=global_addr, [127:126]=type=2
  v4u g0 = { 1u,
             lds,
             (unsigned)ga,
             (unsigned)((ga >> 32) & 0x1FFFFFFu) | (2u << 30) };
  // D# group 1: data_size=4B (code 2) at bits[17:16]; tensor_dim0=n; tensor_dim1=1;
  // tile_dim0=n; tile_dim1=1; tensor_dim0_stride=n; everything else 0.
  v8u g1 = { 0x00020000u,
             (n & 0xFFFFu) << 16,           // tensor_dim0[15:0] -> bits 63:48
             (n >> 16) | (1u << 16),        // tensor_dim0[31:16] | tensor_dim1 lo
             (n & 0xFFFFu) << 16,           // tile_dim0 -> bits 127:112
             1u,                            // tile_dim1 = 1
             n,                             // tensor_dim0_stride lo
             0u, 0u };
  v4u gz = { 0u, 0u, 0u, 0u };
  asm volatile("tensor_load_to_lds %0, %1, %2, %3"
               :: "s"(g0), "s"(g1), "s"(gz), "s"(gz)
               : "memory");
}

__device__ __forceinline__ float pow5f(float x) { float x2 = x * x; return x2 * x2 * x; }
__device__ __forceinline__ float cubef(float x) { return x * x * x; }

// ---------------------------------------------------------------------------
// Kernel 1: metrics. One wave (32 threads) per (b, 16-anchor tile).
// bbox_scores via V_WMMA_F32_16X16X32_F16 one-hot matmul; GT/LUT staged by TDM.
// ---------------------------------------------------------------------------
__device__ __forceinline__ void gather_scores_wmma(const float* arow_base, const int* s_glb,
                                                   float* s_bb, int lane) {
  const int m    = lane & 15;
  const int klo  = (lane < 16) ? 0 : 8;    // A: K sub-block per half-wave
  const int kofs = (lane < 16) ? 0 : 8;    // B: K sub-block per half-wave (same order as A)
  const float* arow = arow_base + (size_t)m * NC;
  for (int g0 = 0; g0 < NG; g0 += 16) {
    v8f acc = {};
    for (int kk = 0; kk < 3; ++kk) {       // classes [0,32), [32,64), [64,96) (padded)
      const int cb = kk * 32;
      v16h av, bv;
#pragma unroll
      for (int j = 0; j < 8; ++j) {
        int c1 = cb + klo + j;             // K = klo..klo+7
        int c2 = cb + 16 + klo + j;        // K = 16+klo..23+klo
        av[j]     = (c1 < NC) ? (_Float16)arow[c1] : (_Float16)0.0f;
        av[j + 8] = (c2 < NC) ? (_Float16)arow[c2] : (_Float16)0.0f;
      }
      const int lbl = s_glb[g0 + m];       // column n = lane&15
#pragma unroll
      for (int j = 0; j < 16; ++j)
        bv[j] = (_Float16)((lbl == cb + kofs + j + (j & 8)) ? 1.0f : 0.0f);
      acc = __builtin_amdgcn_wmma_f32_16x16x32_f16(false, av, false, bv,
                                                   (short)0, acc, false, false);
    }
#pragma unroll
    for (int r = 0; r < 8; ++r)            // D: M = r + 8*(lane>=16), N = lane&15
      s_bb[(g0 + (lane & 15)) * 16 + r + 8 * (lane >> 4)] = acc[r];
  }
}

__global__ __launch_bounds__(32)
void k1_metrics(const float* __restrict__ ps_s, const float* __restrict__ pb_s,
                const float* __restrict__ pa_s, const float* __restrict__ ps_t,
                const float* __restrict__ pb_t, const float* __restrict__ pa_t,
                const float* __restrict__ anc, const int* __restrict__ glab,
                const float* __restrict__ gbox, const float* __restrict__ gang,
                const float* __restrict__ mgt, const float* __restrict__ lidx,
                const float* __restrict__ liou, float* __restrict__ ws,
                float* __restrict__ out) {
  const int b    = blockIdx.y;
  const int a0   = blockIdx.x * 16;
  const int lane = threadIdx.x;

  __shared__ float s_lut[16 * NLUT];
  __shared__ float s_gtb[NG * 4];
  __shared__ int   s_glb[NG];
  __shared__ float s_gan[NG];
  __shared__ float s_mgt[NG];
  __shared__ float s_lidx[16];
  __shared__ int   s_aidx[NG];
  __shared__ float s_gcx[NG], s_gcy[NG], s_gw[NG], s_gh[NG], s_gc[NG], s_gs[NG];
  __shared__ float s_bbs[NG * 16], s_bbt[NG * 16];
  __shared__ float s_pbs[16 * 4], s_pbt[16 * 4], s_pas[16], s_pat[16], s_anc2[16 * 2];

  // --- TDM: stage block-invariant data into LDS (async, TENSORcnt-tracked) ---
  tdm_load_1d(gbox + (size_t)b * NG * 4, lds_offset(s_gtb), NG * 4);
  tdm_load_1d(glab + (size_t)b * NG,     lds_offset(s_glb), NG);
  tdm_load_1d(gang + (size_t)b * NG,     lds_offset(s_gan), NG);
  tdm_load_1d(mgt  + (size_t)b * NG,     lds_offset(s_mgt), NG);
  tdm_load_1d(liou,                      lds_offset(s_lut), 16 * NLUT);

  // Small per-anchor tiles via normal loads (overlap with TDM).
  if (lane < 16) {
    s_lidx[lane] = lidx[lane];
    const int a = a0 + lane;
    const size_t pa4 = ((size_t)b * NA + a) * 4;
#pragma unroll
    for (int k = 0; k < 4; ++k) { s_pbs[lane * 4 + k] = pb_s[pa4 + k];
                                  s_pbt[lane * 4 + k] = pb_t[pa4 + k]; }
    s_pas[lane] = pa_s[(size_t)b * NA + a];
    s_pat[lane] = pa_t[(size_t)b * NA + a];
    s_anc2[lane * 2 + 0] = anc[(size_t)a * 2 + 0];
    s_anc2[lane * 2 + 1] = anc[(size_t)a * 2 + 1];
  }
  __builtin_amdgcn_s_wait_tensorcnt(0);
  __syncthreads();

  // Per-gt precompute (w,h,center,cos/sin,LUT area index).
  for (int g = lane; g < NG; g += 32) {
    const float x1 = s_gtb[g * 4 + 0], y1 = s_gtb[g * 4 + 1];
    const float x2 = s_gtb[g * 4 + 2], y2 = s_gtb[g * 4 + 3];
    const float w = x2 - x1, h = y2 - y1;
    s_gw[g] = w; s_gh[g] = h;
    s_gcx[g] = (x1 + x2) * 0.5f; s_gcy[g] = (y1 + y2) * 0.5f;
    const float th = s_gan[g] * 0.017453292519943295f;
    s_gc[g] = __cosf(th); s_gs[g] = __sinf(th);
    const float ratio = w / (h + 1e-5f);
    int best = 0; float bd = 3.0e38f;
#pragma unroll
    for (int i = 0; i < 16; ++i) {
      const float d = fabsf(ratio - s_lidx[i]);
      if (d < bd) { bd = d; best = i; }
    }
    s_aidx[g] = best;
  }
  __syncthreads();

  // --- WMMA one-hot matmul: bbox_scores (student + teacher) ---
  gather_scores_wmma(ps_s + ((size_t)b * NA + a0) * NC, s_glb, s_bbs, lane);
  gather_scores_wmma(ps_t + ((size_t)b * NA + a0) * NC, s_glb, s_bbt, lane);
  __syncthreads();

  // --- Elementwise metrics over (g, a_local): 48*16 = 768 items / 32 lanes ---
  const size_t rowbase = (size_t)b * NG * NA;
  for (int it = 0; it < 24; ++it) {
    const int idx = it * 32 + lane;
    const int g = idx >> 4, al = idx & 15;
    const int a = a0 + al;

    const float gx1 = s_gtb[g * 4 + 0], gy1 = s_gtb[g * 4 + 1];
    const float gx2 = s_gtb[g * 4 + 2], gy2 = s_gtb[g * 4 + 3];
    const float ag = (gx2 - gx1) * (gy2 - gy1);

    // IoU (student)
    float px1 = s_pbs[al * 4 + 0], py1 = s_pbs[al * 4 + 1];
    float px2 = s_pbs[al * 4 + 2], py2 = s_pbs[al * 4 + 3];
    float iw = fmaxf(fminf(gx2, px2) - fmaxf(gx1, px1), 0.0f);
    float ih = fmaxf(fminf(gy2, py2) - fmaxf(gy1, py1), 0.0f);
    float inter = iw * ih;
    float ap = (px2 - px1) * (py2 - py1);
    const float ovs = inter / (ag + ap - inter + EPSF);
    // IoU (teacher)
    px1 = s_pbt[al * 4 + 0]; py1 = s_pbt[al * 4 + 1];
    px2 = s_pbt[al * 4 + 2]; py2 = s_pbt[al * 4 + 3];
    iw = fmaxf(fminf(gx2, px2) - fmaxf(gx1, px1), 0.0f);
    ih = fmaxf(fminf(gy2, py2) - fmaxf(gy1, py1), 0.0f);
    inter = iw * ih;
    ap = (px2 - px1) * (py2 - py1);
    const float ovt = inter / (ag + ap - inter + EPSF);

    // LUT angle weight
    const int arow = s_aidx[g] * NLUT;
    float td = fminf(fmaxf(fabsf(s_gan[g] - s_pas[al]), 0.0f), 180.0f);
    const float lws = s_lut[arow + (int)(td * (400.0f / 180.0f))];
    td = fminf(fmaxf(fabsf(s_gan[g] - s_pat[al]), 0.0f), 180.0f);
    const float lwt = s_lut[arow + (int)(td * (400.0f / 180.0f))];

    const float bbs = s_bbs[g * 16 + al];
    const float bbt = s_bbt[g * 16 + al];
    const float als = bbs * pow5f(ovs) * cubef(lws);   // ALPHA=1, BETA=5, GAMMA=3
    const float alt = bbt * pow5f(ovt) * cubef(lwt);

    // Rotated inside test
    const float dx = s_anc2[al * 2 + 0] - s_gcx[g];
    const float dy = s_anc2[al * 2 + 1] - s_gcy[g];
    const float xl =  dx * s_gc[g] + dy * s_gs[g];
    const float yl = -dx * s_gs[g] + dy * s_gc[g];
    const float min_ = (fabsf(xl) < s_gw[g] * 0.5f && fabsf(yl) < s_gh[g] * 0.5f) ? 1.0f : 0.0f;

    const size_t o = rowbase + (size_t)g * NA + a;
    ws[W_MET + o] = alt * min_;     // align_mut == align_t (decay term is *0)
    ws[W_AS + o]  = als;
    ws[W_AT + o]  = alt;
    ws[W_OS + o]  = ovs;
    ws[W_OT + o]  = ovt;
    ws[W_MIN + o] = min_;
    out[O_MA + o] = min_ * s_mgt[g];
  }
}

// ---------------------------------------------------------------------------
// Kernel 2: per-(b,g) top-K threshold via binary search on count(> mid).
// ---------------------------------------------------------------------------
__global__ __launch_bounds__(256)
void k2_thresh(float* __restrict__ ws) {
  const int b = blockIdx.y, g = blockIdx.x, t = threadIdx.x;
  const float* row = ws + W_MET + ((size_t)b * NG + g) * NA;
  __shared__ float red[256];
  __shared__ float s_lo, s_hi;
  __shared__ int   s_cnt;

  float mx = 0.0f;
  for (int a = t; a < NA; a += 256) mx = fmaxf(mx, row[a]);
  red[t] = mx; __syncthreads();
  for (int s = 128; s > 0; s >>= 1) {
    if (t < s) red[t] = fmaxf(red[t], red[t + s]);
    __syncthreads();
  }
  if (t == 0) { s_lo = 0.0f; s_hi = red[0]; }
  __syncthreads();

  for (int iter = 0; iter < 24; ++iter) {
    const float mid = 0.5f * (s_lo + s_hi);
    if (t == 0) s_cnt = 0;
    __syncthreads();
    int c = 0;
    for (int a = t; a < NA; a += 256) c += (row[a] > mid) ? 1 : 0;
    atomicAdd(&s_cnt, c);
    __syncthreads();
    if (t == 0) { if (s_cnt > TOPKK) s_lo = mid; else s_hi = mid; }
    __syncthreads();
  }
  if (t == 0) ws[W_THR + (size_t)b * NG + g] = s_hi;
}

// ---------------------------------------------------------------------------
// Kernel 3: per-(b,a) resolution over G: fg / multi / argmax / target gathers.
// ---------------------------------------------------------------------------
__global__ __launch_bounds__(256)
void k3_resolve(const int* __restrict__ glab, const float* __restrict__ gbox,
                const float* __restrict__ gang, const float* __restrict__ mgt,
                float* __restrict__ ws, float* __restrict__ out) {
  const int b = blockIdx.y;
  const int a = blockIdx.x * 256 + threadIdx.x;
  __shared__ float s_thr[NG], s_mgt[NG];
  if (threadIdx.x < NG) {
    s_thr[threadIdx.x] = ws[W_THR + (size_t)b * NG + threadIdx.x];
    s_mgt[threadIdx.x] = mgt[(size_t)b * NG + threadIdx.x];
  }
  __syncthreads();

  const size_t base = (size_t)b * NG * NA + a;
  int fg = 0, firstg = -1, gbest = 0;
  float bestov = -1.0f;
  unsigned long long posbits = 0ull;
  for (int g = 0; g < NG; ++g) {
    const size_t o = base + (size_t)g * NA;
    const float met  = ws[W_MET + o];
    const float ov   = ws[W_OS + o];
    const float min_ = ws[W_MIN + o];
    const bool mp = (met > s_thr[g]) && (met > 0.0f) && (min_ > 0.0f) && (s_mgt[g] > 0.0f);
    if (ov > bestov) { bestov = ov; gbest = g; }
    if (mp) { ++fg; if (firstg < 0) firstg = g; posbits |= (1ull << g); }
  }
  int tg; float fgm;
  if (fg > 1)       { tg = gbest;  fgm = 1.0f; posbits = 1ull << gbest; }
  else if (fg == 1) { tg = firstg; fgm = 1.0f; }
  else              { tg = 0;      fgm = 0.0f; posbits = 0ull; }

  for (int g = 0; g < NG; ++g)
    ws[W_MP + base + (size_t)g * NA] = ((posbits >> g) & 1ull) ? 1.0f : 0.0f;

  const size_t pa = (size_t)b * NA + a;
  const int lbl = glab[(size_t)b * NG + tg];
  out[O_LBL + pa] = (fgm > 0.0f) ? (float)lbl : (float)NC;
  out[O_ANG + pa] = gang[(size_t)b * NG + tg];
  out[O_FG + pa]  = fgm;
#pragma unroll
  for (int k = 0; k < 4; ++k)
    out[O_BOX + pa * 4 + k] = gbox[((size_t)b * NG + tg) * 4 + k];
}

// ---------------------------------------------------------------------------
// Kernel 4: per-(b,g) row maxima (6 values).
// ---------------------------------------------------------------------------
__global__ __launch_bounds__(256)
void k4_rowstats(const float* __restrict__ mgt, float* __restrict__ ws) {
  const int b = blockIdx.y, g = blockIdx.x, t = threadIdx.x;
  const size_t ro = ((size_t)b * NG + g) * NA;
  const float mg = mgt[(size_t)b * NG + g];
  float am = 0, ov = 0, ams = 0, ovs = 0, amt = 0, ovt = 0;
  for (int a = t; a < NA; a += 256) {
    const float as_ = ws[W_AS + ro + a], at_ = ws[W_AT + ro + a];
    const float os_ = ws[W_OS + ro + a], ot_ = ws[W_OT + ro + a];
    const float mi = ws[W_MIN + ro + a] * mg;
    const float mp = ws[W_MP + ro + a];
    am  = fmaxf(am,  as_ * mp); ov  = fmaxf(ov,  os_ * mp);
    ams = fmaxf(ams, as_ * mi); ovs = fmaxf(ovs, os_ * mi);
    amt = fmaxf(amt, at_ * mi); ovt = fmaxf(ovt, ot_ * mi);
  }
  __shared__ float red[256];
  float vals[6] = { am, ov, ams, ovs, amt, ovt };
  for (int k = 0; k < 6; ++k) {
    red[t] = vals[k]; __syncthreads();
    for (int s = 128; s > 0; s >>= 1) {
      if (t < s) red[t] = fmaxf(red[t], red[t + s]);
      __syncthreads();
    }
    if (t == 0) ws[W_RS + ((size_t)b * NG + g) * 6 + k] = red[0];
    __syncthreads();
  }
}

// ---------------------------------------------------------------------------
// Kernel 5: elementwise norm_t / norm_s outputs over (B,G,A).
// ---------------------------------------------------------------------------
__global__ __launch_bounds__(256)
void k5_norm(const float* __restrict__ mgt, const float* __restrict__ ws,
             float* __restrict__ out) {
  const int b = blockIdx.z, g = blockIdx.y;
  const int a = blockIdx.x * 256 + threadIdx.x;
  const size_t o = ((size_t)b * NG + g) * NA + a;
  const float mg = mgt[(size_t)b * NG + g];
  const float* rs = ws + W_RS + ((size_t)b * NG + g) * 6;
  const float ma = ws[W_MIN + o] * mg;
  out[O_NT + o] = ws[W_AT + o] * ma * rs[5] / (rs[4] + EPSF);
  out[O_NS + o] = ws[W_AS + o] * ma * rs[3] / (rs[2] + EPSF);
}

// ---------------------------------------------------------------------------
// Kernel 6: per-(b,a) norm over G + one-hot target_scores.
// ---------------------------------------------------------------------------
__global__ __launch_bounds__(256)
void k6_scores(const float* __restrict__ ws, float* __restrict__ out) {
  const int b = blockIdx.y;
  const int a = blockIdx.x * 256 + threadIdx.x;
  const size_t base = (size_t)b * NG * NA + a;
  float nrm = 0.0f;
  for (int g = 0; g < NG; ++g) {
    const float* rs = ws + W_RS + ((size_t)b * NG + g) * 6;
    const size_t o = base + (size_t)g * NA;
    nrm = fmaxf(nrm, ws[W_AS + o] * ws[W_MP + o] * rs[1] / (rs[0] + EPSF));
  }
  const size_t pa = (size_t)b * NA + a;
  const int lbl = (int)out[O_LBL + pa];
  const float fgm = out[O_FG + pa];
  float* dst = out + O_SCORES + pa * (size_t)NC;
  for (int c = 0; c < NC; ++c) dst[c] = 0.0f;
  if (fgm > 0.0f && lbl >= 0 && lbl < NC) dst[lbl] = nrm;
}

// ---------------------------------------------------------------------------
// Launch
// ---------------------------------------------------------------------------
extern "C" void kernel_launch(void* const* d_in, const int* in_sizes, int n_in,
                              void* d_out, int out_size, void* d_ws, size_t ws_size,
                              hipStream_t stream) {
  const float* ps_s = (const float*)d_in[0];
  const float* pb_s = (const float*)d_in[1];
  const float* pa_s = (const float*)d_in[2];
  const float* ps_t = (const float*)d_in[3];
  const float* pb_t = (const float*)d_in[4];
  const float* pa_t = (const float*)d_in[5];
  const float* anc  = (const float*)d_in[6];
  const int*   glab = (const int*)d_in[7];
  const float* gbox = (const float*)d_in[8];
  const float* gang = (const float*)d_in[9];
  const float* mgt  = (const float*)d_in[10];
  const float* lidx = (const float*)d_in[12];
  const float* liou = (const float*)d_in[13];
  float* out = (float*)d_out;
  float* ws  = (float*)d_ws;

  k1_metrics<<<dim3(NA / 16, BSZ), 32, 0, stream>>>(ps_s, pb_s, pa_s, ps_t, pb_t, pa_t,
                                                    anc, glab, gbox, gang, mgt, lidx,
                                                    liou, ws, out);
  k2_thresh<<<dim3(NG, BSZ), 256, 0, stream>>>(ws);
  k3_resolve<<<dim3(NA / 256, BSZ), 256, 0, stream>>>(glab, gbox, gang, mgt, ws, out);
  k4_rowstats<<<dim3(NG, BSZ), 256, 0, stream>>>(mgt, ws);
  k5_norm<<<dim3(NA / 256, NG, BSZ), 256, 0, stream>>>(mgt, ws, out);
  k6_scores<<<dim3(NA / 256, BSZ), 256, 0, stream>>>(ws, out);
}
